// FrequencyFeatureExtractor_63780264345703
// MI455X (gfx1250) — matrix operand order repaired and checked by reading, not compiled, not run on hardware
//
#include <hip/hip_runtime.h>
#include <hip/hip_bf16.h>

typedef __attribute__((ext_vector_type(16))) _Float16 v16h;
typedef __attribute__((ext_vector_type(8)))  float    v8f;

#define T_STEPS 5
#define NB 8
#define HW 128
#define C1 3
#define C2 64
#define PIX (HW*HW)
#define KTOT (C2*9)          // 576
#define NCHUNK (KTOT/32)     // 18
#define V_TH 0.1f

#if defined(__has_builtin)
#if __has_builtin(__builtin_amdgcn_global_load_async_to_lds_b32) && \
    __has_builtin(__builtin_amdgcn_s_wait_asynccnt)
#define USE_ASYNC_LDS 1
#endif
#endif

typedef __attribute__((address_space(1))) int glb_i32;
typedef __attribute__((address_space(3))) int lds_i32;

__device__ __forceinline__ float clampf(float x, float lo, float hi) {
    return fminf(fmaxf(x, lo), hi);
}

// ---------------------------------------------------------------------------
// K0: pack W2 (f32 OIHW [64][64][3][3]) into f16 A-fragment layout.
// packed[oc_tile(4)][chunk(18)][lane(32)][half(16)], k = r*64 + ic
// 16-bit A 16x32 layout: lane<16 -> M=lane,    K = {0..7,16..23}
//                        lane>=16 -> M=lane-16, K = {8..15,24..31}
// ---------------------------------------------------------------------------
__global__ void pack_w2_kernel(const float* __restrict__ W2,
                               _Float16* __restrict__ wp) {
    int idx = blockIdx.x * blockDim.x + threadIdx.x;     // 4*18*32*16 = 36864
    if (idx >= 4 * NCHUNK * 32 * 16) return;
    int h     = idx & 15;
    int lane  = (idx >> 4) & 31;
    int chunk = (idx >> 9) % NCHUNK;
    int tile  = idx / (16 * 32 * NCHUNK);
    int K;
    if (lane < 16) K = (h < 8) ? h       : (h + 8);   // 0..7, 16..23
    else           K = (h < 8) ? (h + 8) : (h + 16);  // 8..15, 24..31
    int M  = lane & 15;
    int oc = tile * 16 + M;
    int k  = chunk * 32 + K;
    int r  = k >> 6;       // tap index 0..8
    int ic = k & 63;
    wp[idx] = (_Float16)W2[(oc * C2 + ic) * 9 + r];
}

// ---------------------------------------------------------------------------
// K1: conv1 (3->64, 3x3, pad 1) + LIF1 (input repeats across T -> conv once,
// 5-step recurrence in registers). Writes f16 spikes s1[t][b][ic][y][x].
// ---------------------------------------------------------------------------
__global__ void conv1_lif1_kernel(const float* __restrict__ in,
                                  const float* __restrict__ W1,
                                  const float* __restrict__ b1,
                                  const float* __restrict__ tau_p,
                                  _Float16* __restrict__ s1) {
    __shared__ float w1s[64 * C1 * 9];                   // 6912 B
    for (int i = threadIdx.x; i < 64 * C1 * 9; i += blockDim.x)
        w1s[i] = W1[i];
    __syncthreads();

    int gid = blockIdx.x * blockDim.x + threadIdx.x;     // NB*PIX = 131072
    if (gid >= NB * PIX) return;
    int b = gid / PIX;
    int p = gid % PIX;
    int y = p / HW, x = p % HW;

    float tau = clampf(*tau_p, 0.5f, 5.0f);
    float inv = 1.0f / tau, keep = 1.0f - inv;

    float xv[C1 * 9];
#pragma unroll
    for (int c = 0; c < C1; ++c)
#pragma unroll
        for (int dy = 0; dy < 3; ++dy)
#pragma unroll
            for (int dx = 0; dx < 3; ++dx) {
                int gy = y + dy - 1, gx = x + dx - 1;
                float v = 0.0f;
                if (gy >= 0 && gy < HW && gx >= 0 && gx < HW)
                    v = in[((b * C1 + c) * HW + gy) * HW + gx];
                xv[(c * 3 + dy) * 3 + dx] = v;
            }

    for (int oc = 0; oc < C2; ++oc) {
        float acc = b1[oc];
        const float* w = &w1s[oc * C1 * 9];
#pragma unroll
        for (int j = 0; j < C1 * 9; ++j) acc = fmaf(w[j], xv[j], acc);
        float v = 0.0f;
#pragma unroll
        for (int t = 0; t < T_STEPS; ++t) {
            v = v * keep + inv * acc;
            float s = (v >= V_TH) ? 1.0f : 0.0f;
            v = (s > 0.0f) ? 0.0f : v;
            v = clampf(v, -2.0f, 2.0f);
            s1[(((size_t)t * NB + b) * C2 + oc) * PIX + p] = (_Float16)s;
        }
    }
}

// ---------------------------------------------------------------------------
// K2: conv2 (64->64, 3x3) as implicit GEMM on v_wmma_f32_16x16x32_f16,
// fused with LIF2. Block: 256 thr = 8 waves = 4 oc-tiles x 2 pixel-tiles
// over a 32-pixel row strip. A-fragments (weights) are re-read from wp
// per chunk in every t step; B halo tile staged in LDS via async global->LDS.
// ---------------------------------------------------------------------------
#define TILE_COLS 34
#define ROW_ELEMS (3 * TILE_COLS)                        // 102
#define TILE_ELEMS (C2 * ROW_ELEMS)                      // 6528 halves

__global__ void __launch_bounds__(256)
conv2_lif2_wmma_kernel(const _Float16* __restrict__ s1,
                       const _Float16* __restrict__ wp,
                       const float* __restrict__ b2,
                       const float* __restrict__ tau_p,
                       _Float16* __restrict__ s2) {
    __shared__ __attribute__((aligned(16))) _Float16 tile[TILE_ELEMS]; // 13056 B

    const int wave = threadIdx.x >> 5;
    const int lane = threadIdx.x & 31;
    const int oc_tile = wave >> 1;                       // 0..3
    const int n_tile  = wave & 1;                        // 0..1
    const int x0 = blockIdx.x * 32;                      // strip base
    const int y  = blockIdx.y;
    const int b  = blockIdx.z;

    float tau = clampf(*tau_p, 0.5f, 5.0f);
    float inv = 1.0f / tau, keep = 1.0f - inv;

    v8f vstate = {};                                     // LIF2 membrane
    for (int t = 0; t < T_STEPS; ++t) {
        __syncthreads();                                 // protect prev reads
        const size_t tb = ((size_t)t * NB + b) * C2;

        // ---- stage spike halo tile: async DMA for interior, zeros for pad
        for (int pr = threadIdx.x; pr < TILE_ELEMS / 2; pr += blockDim.x) {
            int ic  = pr / (3 * 17);
            int rem = pr % (3 * 17);
            int row = rem / 17;
            int col = (rem % 17) * 2;                    // even col of the pair
            int gy = y + row - 1, gx = x0 + col - 1;
            bool rowok = (gy >= 0 && gy < HW);
            bool v0 = rowok && (gx >= 0 && gx < HW);
            bool v1 = rowok && (gx + 1 >= 0 && gx + 1 < HW);
            int di = ic * ROW_ELEMS + row * TILE_COLS + col;
#if defined(USE_ASYNC_LDS)
            if (v0 && v1) {
                __builtin_amdgcn_global_load_async_to_lds_b32(
                    (glb_i32*)(uintptr_t)(s1 + (tb + ic) * PIX + gy * HW + gx),
                    (lds_i32*)(uintptr_t)(&tile[di]),
                    0, 0);
            } else {
                tile[di]     = v0 ? s1[(tb + ic) * PIX + gy * HW + gx]
                                  : (_Float16)0.0f;
                tile[di + 1] = v1 ? s1[(tb + ic) * PIX + gy * HW + gx + 1]
                                  : (_Float16)0.0f;
            }
#else
            tile[di]     = v0 ? s1[(tb + ic) * PIX + gy * HW + gx]
                              : (_Float16)0.0f;
            tile[di + 1] = v1 ? s1[(tb + ic) * PIX + gy * HW + gx + 1]
                              : (_Float16)0.0f;
#endif
        }
#if defined(USE_ASYNC_LDS)
        __builtin_amdgcn_s_wait_asynccnt(0);
#endif
        __syncthreads();

        // ---- 18 x v_wmma_f32_16x16x32_f16. sched_barrier(DS-read allowed)
        //      caps A/B fragments in flight (no scratch spills) while letting
        //      the next chunk's LDS loads issue under the current WMMA.
        v8f c = {};
#pragma unroll
        for (int chunk = 0; chunk < NCHUNK; ++chunk) {
            v16h a = *(const v16h*)(wp +
                (((size_t)oc_tile * NCHUNK + chunk) * 32 + lane) * 16);
            // tap index is uniform per chunk: k = r*64+ic, chunk covers 32 ics
            const int r  = chunk >> 1;                   // 0..8
            const int dy = r / 3, dx = r % 3;
            const int ic = ((chunk & 1) << 5);           // chunk ic base; lane <-> N col, elem <-> K (table order)
            const _Float16* src = &tile[ic * ROW_ELEMS + dy * TILE_COLS
                                        + n_tile * 16 + (lane & 15) + dx];
            v16h bf;
#pragma unroll
            for (int j = 0; j < 16; ++j) bf[j] = src[((j & 7) + ((j >> 3) << 4) + ((lane >> 4) << 3)) * ROW_ELEMS];
            c = __builtin_amdgcn_wmma_f32_16x16x32_f16(
                    false, a, false, bf, (short)0, c, false, false);
            if (chunk & 1) __builtin_amdgcn_sched_barrier(0x100); // DS reads may cross
        }

        // ---- fused LIF2 on the 16x16 f32 accumulator tile
        const int N = lane & 15;
        const int x = x0 + n_tile * 16 + N;
#pragma unroll
        for (int i = 0; i < 8; ++i) {
            int M  = i + ((lane >= 16) ? 8 : 0);
            int oc = oc_tile * 16 + M;
            float xin = c[i] + b2[oc];
            float v = vstate[i] * keep + inv * xin;
            float s = (v >= V_TH) ? 1.0f : 0.0f;
            v = (s > 0.0f) ? 0.0f : v;
            vstate[i] = clampf(v, -2.0f, 2.0f);
            s2[(tb + oc) * PIX + y * HW + x] = (_Float16)s;
        }
    }
}

// ---------------------------------------------------------------------------
// K3: head 1x1 conv (64->3), memory-bound GEMV per pixel.
// ---------------------------------------------------------------------------
__global__ void head_conv_kernel(const _Float16* __restrict__ s2,
                                 const float* __restrict__ Wh,
                                 const float* __restrict__ bh,
                                 float* __restrict__ out) {
    __shared__ float whs[3 * C2];
    for (int i = threadIdx.x; i < 3 * C2; i += blockDim.x) whs[i] = Wh[i];
    __syncthreads();

    int gid = blockIdx.x * blockDim.x + threadIdx.x;     // T*NB*PIX = 655360
    if (gid >= T_STEPS * NB * PIX) return;
    int p  = gid % PIX;
    int tb = gid / PIX;                                  // t*NB + b

    float acc0 = bh[0], acc1 = bh[1], acc2 = bh[2];
    const _Float16* sp = s2 + (size_t)tb * C2 * PIX + p;
#pragma unroll 8
    for (int ic = 0; ic < C2; ++ic) {
        float f = (float)sp[(size_t)ic * PIX];
        acc0 = fmaf(whs[0 * C2 + ic], f, acc0);
        acc1 = fmaf(whs[1 * C2 + ic], f, acc1);
        acc2 = fmaf(whs[2 * C2 + ic], f, acc2);
    }
    float* op = out + (size_t)tb * 3 * PIX + p;
    op[0 * PIX] = acc0;
    op[1 * PIX] = acc1;
    op[2 * PIX] = acc2;
}

// ---------------------------------------------------------------------------
extern "C" void kernel_launch(void* const* d_in, const int* in_sizes, int n_in,
                              void* d_out, int out_size, void* d_ws, size_t ws_size,
                              hipStream_t stream) {
    const float* lr_lab = (const float*)d_in[0];
    const float* W1     = (const float*)d_in[1];
    const float* b1     = (const float*)d_in[2];
    const float* tau1   = (const float*)d_in[3];
    const float* W2     = (const float*)d_in[4];
    const float* b2     = (const float*)d_in[5];
    const float* tau2   = (const float*)d_in[6];
    const float* Wh     = (const float*)d_in[7];
    const float* bh     = (const float*)d_in[8];
    float* out = (float*)d_out;

    char* ws = (char*)d_ws;
    const size_t s1_elems = (size_t)T_STEPS * NB * C2 * PIX;   // 41.9M halves
    _Float16* s1 = (_Float16*)ws;                       ws += s1_elems * sizeof(_Float16);
    _Float16* s2 = (_Float16*)ws;                       ws += s1_elems * sizeof(_Float16);
    _Float16* wp = (_Float16*)ws;                       // 36864 halves

    pack_w2_kernel<<<(4 * NCHUNK * 32 * 16 + 255) / 256, 256, 0, stream>>>(W2, wp);

    conv1_lif1_kernel<<<(NB * PIX + 255) / 256, 256, 0, stream>>>(
        lr_lab, W1, b1, tau1, s1);

    dim3 g2(HW / 32, HW, NB);                            // (4, 128, 8)
    conv2_lif2_wmma_kernel<<<g2, 256, 0, stream>>>(s1, wp, b2, tau2, s2);

    head_conv_kernel<<<(T_STEPS * NB * PIX + 255) / 256, 256, 0, stream>>>(
        s2, Wh, bh, out);
}
